// Mymodel_GAT_residue_89051851915806
// MI455X (gfx1250) — matrix operand order repaired and checked by reading, not compiled, not run on hardware
//
#include <hip/hip_runtime.h>

typedef __attribute__((ext_vector_type(16))) _Float16 v16h;
typedef __attribute__((ext_vector_type(8)))  float    v8f;

#define NN    30000
#define NE    300000
#define DIN   64
#define HIDC  128
#define NHEAD 4
#define EDGC  256
#define OUTC  86
#define OUTP  96
#define EPSV  1e-5f
#define NEG_SLOPE 0.2f

union FragHalf { v16h h; float4 q[2]; };

// ---------------------------------------------------------------------------
// Generic WMMA GEMM:  C[M,N] = A[M,K] (f16) @ W[N,K]^T (f16)  (+bias)
// One wave computes a 16 x (NT*16) strip; the block's shared W panel is
// double-buffered through LDS with global_load_async_to_lds_b128 (ASYNCcnt).
// GATHER mode builds A rows on the fly from xr[src[e]] / xr[dst[e]].
// ---------------------------------------------------------------------------
template <int NT, bool GATHER, typename OUT_T, bool BIAS, bool NGUARD>
__global__ void wmma_gemm(const _Float16* __restrict__ Amat, int lda,
                          const int* __restrict__ gsrc, const int* __restrict__ gdst,
                          const _Float16* __restrict__ Xg,
                          const _Float16* __restrict__ W, int ldw,
                          const float* __restrict__ bias,
                          OUT_T* __restrict__ C, int ldc, int Ncols,
                          int Mtiles, int K)
{
    // double-buffered W panel: up to 64 rows x 32 halves, 80B row pitch
    // (pitch 80B => 16 consecutive rows start in distinct bank groups)
    __shared__ _Float16 bsm[2][64][40];

    const int wave = threadIdx.x >> 5;
    int mtile = blockIdx.x * (blockDim.x >> 5) + wave;
    const bool mvalid = (mtile < Mtiles);
    if (!mvalid) mtile = Mtiles - 1;        // clamp: stay in barriers, loads in-bounds
    const int lane = threadIdx.x & 31;
    const int mrow = lane & 15;             // A row within tile / B column
    const int half = lane >> 4;
    const int n0   = blockIdx.y * (NT * 16);

    // cooperative async loader: NT*64 lanes move NT*16 rows x 64B per K-step
    const bool loader = ((int)threadIdx.x) < NT * 64;   // whole waves (NT*64 % 32 == 0)
    const int  lrow   = threadIdx.x >> 2;
    const int  lchunk = threadIdx.x & 3;
    const _Float16* gB  = W + (size_t)(n0 + lrow) * ldw + lchunk * 8;
    const unsigned lds0 = (unsigned)(uintptr_t)&bsm[0][lrow][lchunk * 8];
    const unsigned lds1 = (unsigned)(uintptr_t)&bsm[1][lrow][lchunk * 8];

    v8f acc[NT];
#pragma unroll
    for (int t = 0; t < NT; ++t) acc[t] = v8f{};

    const _Float16* arowS;
    const _Float16* arowD = nullptr;
    if (GATHER) {
        const int e = mtile * 16 + mrow;
        arowS = Xg + (long long)gsrc[e] * HIDC;
        arowD = Xg + (long long)gdst[e] * HIDC;
    } else {
        arowS = Amat + (long long)(mtile * 16 + mrow) * lda;
    }

    const int nsteps = K / 32;
    if (loader)
        asm volatile("global_load_async_to_lds_b128 %0, %1, off"
                     :: "v"(lds0), "v"(gB) : "memory");

    for (int ks = 0; ks < nsteps; ++ks) {
        const int cur = ks & 1;
        if (ks + 1 < nsteps) {
            if (loader) {
                const unsigned dstl = cur ? lds0 : lds1;
                asm volatile("global_load_async_to_lds_b128 %0, %1, off"
                             :: "v"(dstl), "v"(gB + (size_t)(ks + 1) * 32) : "memory");
            }
            asm volatile("s_wait_asynccnt 0x1" ::: "memory");   // current panel landed
        } else {
            asm volatile("s_wait_asynccnt 0x0" ::: "memory");
        }
        __syncthreads();

        // ---- A fragment: 16x32 f16 (ISA 7.12.2 16-bit A layout) ----
        const int k0 = ks * 32;
        const _Float16* ab;
        if (GATHER) ab = (k0 < HIDC) ? (arowS + k0) : (arowD + (k0 - HIDC));
        else        ab = arowS + k0;
        FragHalf a;
        a.q[0] = *reinterpret_cast<const float4*>(ab + half * 8);
        a.q[1] = *reinterpret_cast<const float4*>(ab + 16 + half * 8);

#pragma unroll
        for (int t = 0; t < NT; ++t) {
            // ---- B fragment from LDS: col n = lane&15 reads row, K in A's table order ----
            const _Float16* bb = &bsm[cur][t * 16 + mrow][half * 8];
            FragHalf b;
            b.q[0] = *reinterpret_cast<const float4*>(bb);
            b.q[1] = *reinterpret_cast<const float4*>(bb + 16);
            acc[t] = __builtin_amdgcn_wmma_f32_16x16x32_f16(
                         false, a.h, false, b.h, (short)0, acc[t], false, false);
        }
        __syncthreads();   // protect buf[cur] before it is overwritten at ks+1's issue
    }

#pragma unroll
    for (int t = 0; t < NT; ++t) {
        const int col = n0 + t * 16 + mrow;
        float bv = 0.0f;
        if (BIAS) bv = bias[(NGUARD && col >= Ncols) ? 0 : col];
#pragma unroll
        for (int r = 0; r < 8; ++r) {
            const long long row = (long long)mtile * 16 + half * 8 + r;
            const float v = acc[t][r] + bv;
            if (mvalid && (!NGUARD || col < Ncols))
                C[row * ldc + col] = (OUT_T)v;
        }
    }
}

// ---------------------------------------------------------------------------
// Elementwise / reduction helpers
// ---------------------------------------------------------------------------
__global__ void f32_to_f16_k(const float* __restrict__ in, _Float16* __restrict__ out, long long n)
{
    long long t = (long long)blockIdx.x * blockDim.x + threadIdx.x;
    if (t < n) out[t] = (_Float16)in[t];
}

__global__ void cvt_pad_w8_k(const float* __restrict__ in, _Float16* __restrict__ out)
{
    int t = blockIdx.x * blockDim.x + threadIdx.x;   // OUTP*EDGC threads
    if (t >= OUTP * EDGC) return;
    int row = t / EDGC, col = t % EDGC;
    out[t] = (row < OUTC) ? (_Float16)in[row * EDGC + col] : (_Float16)0.0f;
}

// a_s[n,h] = sum_c h[n,h*128+c]*ats[h*128+c]   (block=128 threads, 1 node/block)
__global__ void attn_scores_k(const float* __restrict__ h,
                              const float* __restrict__ ats, const float* __restrict__ atd,
                              float* __restrict__ a_s, float* __restrict__ a_d)
{
    const int n = blockIdx.x;
    const int c = threadIdx.x;
    const float* hr = h + (long long)n * (NHEAD * HIDC);
    float s[NHEAD], d[NHEAD];
#pragma unroll
    for (int hh = 0; hh < NHEAD; ++hh) {
        float v = hr[hh * HIDC + c];
        s[hh] = v * ats[hh * HIDC + c];
        d[hh] = v * atd[hh * HIDC + c];
    }
#pragma unroll
    for (int o = 16; o > 0; o >>= 1)
#pragma unroll
        for (int hh = 0; hh < NHEAD; ++hh) {
            s[hh] += __shfl_down(s[hh], o, 32);
            d[hh] += __shfl_down(d[hh], o, 32);
        }
    __shared__ float ps[4][NHEAD], pd[4][NHEAD];
    const int wave = threadIdx.x >> 5;
    if ((threadIdx.x & 31) == 0)
#pragma unroll
        for (int hh = 0; hh < NHEAD; ++hh) { ps[wave][hh] = s[hh]; pd[wave][hh] = d[hh]; }
    __syncthreads();
    if (threadIdx.x < NHEAD) {
        float ss = 0.f, dd = 0.f;
        for (int w = 0; w < 4; ++w) { ss += ps[w][threadIdx.x]; dd += pd[w][threadIdx.x]; }
        a_s[n * NHEAD + threadIdx.x] = ss;
        a_d[n * NHEAD + threadIdx.x] = dd;
    }
}

__device__ __forceinline__ unsigned f2ord(float x) {
    unsigned u = __float_as_uint(x);
    return (u & 0x80000000u) ? ~u : (u | 0x80000000u);
}
__device__ __forceinline__ float ord2f(unsigned u) {
    return (u & 0x80000000u) ? __uint_as_float(u & 0x7fffffffu) : __uint_as_float(~u);
}

__global__ void edge_alpha_k(const float* __restrict__ a_s, const float* __restrict__ a_d,
                             const int* __restrict__ src, const int* __restrict__ dst,
                             float* __restrict__ al, unsigned* __restrict__ amax)
{
    long long tid = (long long)blockIdx.x * blockDim.x + threadIdx.x;
    if (tid >= (long long)NE * NHEAD) return;
    const int e = (int)(tid >> 2), hh = (int)(tid & 3);
    float a = a_s[src[e] * NHEAD + hh] + a_d[dst[e] * NHEAD + hh];
    a = (a > 0.f) ? a : a * NEG_SLOPE;
    al[tid] = a;
    atomicMax(&amax[dst[e] * NHEAD + hh], f2ord(a));
}

__global__ void edge_exp_k(float* __restrict__ al, const unsigned* __restrict__ amax,
                           const int* __restrict__ dst, float* __restrict__ denom)
{
    long long tid = (long long)blockIdx.x * blockDim.x + threadIdx.x;
    if (tid >= (long long)NE * NHEAD) return;
    const int e = (int)(tid >> 2), hh = (int)(tid & 3);
    float ex = __expf(al[tid] - ord2f(amax[dst[e] * NHEAD + hh]));
    al[tid] = ex;
    unsafeAtomicAdd(&denom[dst[e] * NHEAD + hh], ex);
}

__global__ void edge_coef_k(float* __restrict__ al, const float* __restrict__ denom,
                            const int* __restrict__ dst)
{
    long long tid = (long long)blockIdx.x * blockDim.x + threadIdx.x;
    if (tid >= (long long)NE * NHEAD) return;
    const int e = (int)(tid >> 2), hh = (int)(tid & 3);
    al[tid] = al[tid] / denom[dst[e] * NHEAD + hh] * (1.0f / NHEAD);   // fold head-mean
}

// agg[dst,c] += sum_h h[src, h*128+c] * coef[e,h]     (block=256 -> 2 edges)
__global__ void aggregate_k(const float* __restrict__ h, const float* __restrict__ cf,
                            const int* __restrict__ src, const int* __restrict__ dst,
                            float* __restrict__ agg)
{
    const int e = blockIdx.x * 2 + (threadIdx.x >> 7);
    const int c = threadIdx.x & 127;
    if (e >= NE) return;
    const int s = src[e], d = dst[e];
    const float* hr = h + (long long)s * (NHEAD * HIDC);
    const float c0 = cf[e * 4 + 0], c1 = cf[e * 4 + 1], c2 = cf[e * 4 + 2], c3 = cf[e * 4 + 3];
    float v = hr[c] * c0 + hr[HIDC + c] * c1 + hr[2 * HIDC + c] * c2 + hr[3 * HIDC + c] * c3;
    unsafeAtomicAdd(&agg[(long long)d * HIDC + c], v);
}

template <typename T>
__global__ void col_stats_k(const T* __restrict__ X, long long M, int N,
                            float* __restrict__ sum, float* __restrict__ sumsq)
{
    const int c = threadIdx.x;           // blockDim.x == N
    float s = 0.f, q = 0.f;
    for (long long r = blockIdx.x; r < M; r += gridDim.x) {
        float v = (float)X[r * N + c];
        s += v; q += v * v;
    }
    unsafeAtomicAdd(&sum[c], s);
    unsafeAtomicAdd(&sumsq[c], q);
}

__global__ void bn_relu_node_k(const float* __restrict__ pre, const float* __restrict__ sum,
                               const float* __restrict__ sumsq, const float* __restrict__ w,
                               const float* __restrict__ b, const float* __restrict__ resid,
                               float* __restrict__ out32, _Float16* __restrict__ out16)
{
    long long tid = (long long)blockIdx.x * blockDim.x + threadIdx.x;
    if (tid >= (long long)NN * HIDC) return;
    const int c = (int)(tid & (HIDC - 1));
    const float inv = 1.0f / NN;
    const float m = sum[c] * inv;
    const float var = sumsq[c] * inv - m * m;
    float y = (pre[tid] - m) * rsqrtf(var + EPSV) * w[c] + b[c];
    if (resid) y += resid[tid];
    y = fmaxf(y, 0.f);
    out32[tid] = y;
    out16[tid] = (_Float16)y;
}

// y = relu( BN(T)[e,c] + resid )  ; resid = A buffer, or gathered concat(xr[src],xr[dst])
__global__ void bn_edge_k(const _Float16* __restrict__ T, const float* __restrict__ sum,
                          const float* __restrict__ sumsq, const float* __restrict__ w,
                          const float* __restrict__ b, const int* __restrict__ src,
                          const int* __restrict__ dst, const float* __restrict__ xr,
                          const _Float16* __restrict__ residA, _Float16* __restrict__ out)
{
    long long tid = (long long)blockIdx.x * blockDim.x + threadIdx.x;
    if (tid >= (long long)NE * EDGC) return;
    const int c = (int)(tid & (EDGC - 1));
    const long long e = tid >> 8;
    const float inv = 1.0f / (float)NE;
    const float m = sum[c] * inv;
    const float var = sumsq[c] * inv - m * m;
    float y = ((float)T[tid] - m) * rsqrtf(var + EPSV) * w[c] + b[c];
    float r;
    if (residA) r = (float)residA[tid];
    else {
        const int node = (c < HIDC) ? src[e] : dst[e];
        r = xr[(long long)node * HIDC + (c & (HIDC - 1))];
    }
    y = fmaxf(y + r, 0.f);
    out[tid] = (_Float16)y;
}

// ---------------------------------------------------------------------------
extern "C" void kernel_launch(void* const* d_in, const int* in_sizes, int n_in,
                              void* d_out, int out_size, void* d_ws, size_t ws_size,
                              hipStream_t stream)
{
    const float* x    = (const float*)d_in[0];
    const int*   ei   = (const int*)d_in[1];
    const int*   src  = ei;
    const int*   dst  = ei + NE;
    const float* w0   = (const float*)d_in[2];
    const float* ats0 = (const float*)d_in[4];
    const float* atd0 = (const float*)d_in[5];
    const float* bn1w = (const float*)d_in[6];
    const float* bn1b = (const float*)d_in[7];
    const float* w2   = (const float*)d_in[8];
    const float* ats1 = (const float*)d_in[10];
    const float* atd1 = (const float*)d_in[11];
    const float* bn2w = (const float*)d_in[12];
    const float* bn2b = (const float*)d_in[13];
    const float* w4   = (const float*)d_in[14];
    const float* bn3w = (const float*)d_in[16];
    const float* bn3b = (const float*)d_in[17];
    const float* w6   = (const float*)d_in[18];
    const float* bn4w = (const float*)d_in[20];
    const float* bn4b = (const float*)d_in[21];
    const float* w8   = (const float*)d_in[22];
    const float* b8   = (const float*)d_in[23];
    float* out = (float*)d_out;

    // ---- workspace carve-up (256B aligned) ----
    char*  ws  = (char*)d_ws;
    size_t off = 0;
    auto carve = [&](size_t bytes) -> char* {
        char* p = ws + off;
        off = (off + bytes + 255) & ~(size_t)255;
        return p;
    };
    float*     hbuf  = (float*)carve((size_t)NN * 512 * 4);
    float*     a_s   = (float*)carve((size_t)NN * 4 * 4);
    float*     a_d   = (float*)carve((size_t)NN * 4 * 4);
    unsigned*  amax  = (unsigned*)carve((size_t)NN * 4 * 4);
    float*     denom = (float*)carve((size_t)NN * 4 * 4);
    float*     albuf = (float*)carve((size_t)NE * 4 * 4);
    float*     agg   = (float*)carve((size_t)NN * HIDC * 4);
    float*     x1f   = (float*)carve((size_t)NN * HIDC * 4);
    float*     xrf   = (float*)carve((size_t)NN * HIDC * 4);
    _Float16*  x1h   = (_Float16*)carve((size_t)NN * HIDC * 2);
    _Float16*  xrh   = (_Float16*)carve((size_t)NN * HIDC * 2);
    _Float16*  xh    = (_Float16*)carve((size_t)NN * DIN * 2);
    _Float16*  w0h   = (_Float16*)carve((size_t)512 * DIN * 2);
    _Float16*  w2h   = (_Float16*)carve((size_t)512 * HIDC * 2);
    _Float16*  w4h   = (_Float16*)carve((size_t)EDGC * EDGC * 2);
    _Float16*  w6h   = (_Float16*)carve((size_t)EDGC * EDGC * 2);
    _Float16*  w8h   = (_Float16*)carve((size_t)OUTP * EDGC * 2);
    float*     csum  = (float*)carve(256 * 4);
    float*     csq   = (float*)carve(256 * 4);
    _Float16*  Tbuf  = (_Float16*)carve((size_t)NE * EDGC * 2);
    _Float16*  Abuf  = (_Float16*)carve((size_t)NE * EDGC * 2);

    const long long E4 = (long long)NE * NHEAD;
    const int EW = 256;
    auto gridFor = [](long long n, int b) { return (unsigned)((n + b - 1) / b); };

    // ---- 0: convert inputs/weights to f16 (w8 padded to 96 rows) ----
    f32_to_f16_k<<<gridFor((long long)NN * DIN, EW), EW, 0, stream>>>(x, xh, (long long)NN * DIN);
    f32_to_f16_k<<<gridFor(512 * DIN, EW), EW, 0, stream>>>(w0, w0h, 512 * DIN);
    f32_to_f16_k<<<gridFor(512 * HIDC, EW), EW, 0, stream>>>(w2, w2h, 512 * HIDC);
    f32_to_f16_k<<<gridFor(EDGC * EDGC, EW), EW, 0, stream>>>(w4, w4h, EDGC * EDGC);
    f32_to_f16_k<<<gridFor(EDGC * EDGC, EW), EW, 0, stream>>>(w6, w6h, EDGC * EDGC);
    cvt_pad_w8_k<<<gridFor(OUTP * EDGC, EW), EW, 0, stream>>>(w8, w8h);

    const int MT_N = NN / 16;    // 1875
    const int MT_E = NE / 16;    // 18750
    dim3 blk(256);

    // ================= GAT layer 1 =================
    {   // h = x @ w0^T   [NN,512], K=64
        dim3 g((MT_N + 7) / 8, 512 / 64);
        wmma_gemm<4, false, float, false, false><<<g, blk, 0, stream>>>(
            xh, DIN, nullptr, nullptr, nullptr, w0h, DIN, nullptr, hbuf, 512, 512, MT_N, DIN);
    }
    attn_scores_k<<<NN, HIDC, 0, stream>>>(hbuf, ats0, atd0, a_s, a_d);
    hipMemsetAsync(amax, 0, (size_t)NN * 4 * 4, stream);
    hipMemsetAsync(denom, 0, (size_t)NN * 4 * 4, stream);
    hipMemsetAsync(agg, 0, (size_t)NN * HIDC * 4, stream);
    edge_alpha_k<<<gridFor(E4, EW), EW, 0, stream>>>(a_s, a_d, src, dst, albuf, amax);
    edge_exp_k<<<gridFor(E4, EW), EW, 0, stream>>>(albuf, amax, dst, denom);
    edge_coef_k<<<gridFor(E4, EW), EW, 0, stream>>>(albuf, denom, dst);
    aggregate_k<<<NE / 2, 256, 0, stream>>>(hbuf, albuf, src, dst, agg);
    hipMemsetAsync(csum, 0, 256 * 4, stream);
    hipMemsetAsync(csq, 0, 256 * 4, stream);
    col_stats_k<float><<<512, HIDC, 0, stream>>>(agg, NN, HIDC, csum, csq);
    bn_relu_node_k<<<gridFor((long long)NN * HIDC, EW), EW, 0, stream>>>(
        agg, csum, csq, bn1w, bn1b, nullptr, x1f, x1h);

    // ================= GAT layer 2 =================
    {   // h = x1 @ w2^T  [NN,512], K=128
        dim3 g((MT_N + 7) / 8, 512 / 64);
        wmma_gemm<4, false, float, false, false><<<g, blk, 0, stream>>>(
            x1h, HIDC, nullptr, nullptr, nullptr, w2h, HIDC, nullptr, hbuf, 512, 512, MT_N, HIDC);
    }
    attn_scores_k<<<NN, HIDC, 0, stream>>>(hbuf, ats1, atd1, a_s, a_d);
    hipMemsetAsync(amax, 0, (size_t)NN * 4 * 4, stream);
    hipMemsetAsync(denom, 0, (size_t)NN * 4 * 4, stream);
    hipMemsetAsync(agg, 0, (size_t)NN * HIDC * 4, stream);
    edge_alpha_k<<<gridFor(E4, EW), EW, 0, stream>>>(a_s, a_d, src, dst, albuf, amax);
    edge_exp_k<<<gridFor(E4, EW), EW, 0, stream>>>(albuf, amax, dst, denom);
    edge_coef_k<<<gridFor(E4, EW), EW, 0, stream>>>(albuf, denom, dst);
    aggregate_k<<<NE / 2, 256, 0, stream>>>(hbuf, albuf, src, dst, agg);
    hipMemsetAsync(csum, 0, 256 * 4, stream);
    hipMemsetAsync(csq, 0, 256 * 4, stream);
    col_stats_k<float><<<512, HIDC, 0, stream>>>(agg, NN, HIDC, csum, csq);
    bn_relu_node_k<<<gridFor((long long)NN * HIDC, EW), EW, 0, stream>>>(
        agg, csum, csq, bn2w, bn2b, x1f, xrf, xrh);

    // ================= edge MLP =================
    {   // T = concat(xr[src],xr[dst]) @ w4^T   (gathered A), f16 out
        dim3 g((MT_E + 7) / 8, EDGC / 64);
        wmma_gemm<4, true, _Float16, false, false><<<g, blk, 0, stream>>>(
            nullptr, 0, src, dst, xrh, w4h, EDGC, nullptr, Tbuf, EDGC, EDGC, MT_E, EDGC);
    }
    hipMemsetAsync(csum, 0, 256 * 4, stream);
    hipMemsetAsync(csq, 0, 256 * 4, stream);
    col_stats_k<_Float16><<<1024, EDGC, 0, stream>>>(Tbuf, NE, EDGC, csum, csq);
    bn_edge_k<<<gridFor((long long)NE * EDGC, EW), EW, 0, stream>>>(
        Tbuf, csum, csq, bn3w, bn3b, src, dst, xrf, nullptr, Abuf);

    {   // T = A @ w6^T, f16 out
        dim3 g((MT_E + 7) / 8, EDGC / 64);
        wmma_gemm<4, false, _Float16, false, false><<<g, blk, 0, stream>>>(
            Abuf, EDGC, nullptr, nullptr, nullptr, w6h, EDGC, nullptr, Tbuf, EDGC, EDGC, MT_E, EDGC);
    }
    hipMemsetAsync(csum, 0, 256 * 4, stream);
    hipMemsetAsync(csq, 0, 256 * 4, stream);
    col_stats_k<_Float16><<<1024, EDGC, 0, stream>>>(Tbuf, NE, EDGC, csum, csq);
    bn_edge_k<<<gridFor((long long)NE * EDGC, EW), EW, 0, stream>>>(
        Tbuf, csum, csq, bn4w, bn4b, src, dst, nullptr, Abuf, Tbuf);   // in-place elementwise

    {   // out = T @ w8^T + b8   [NE,86], padded N=96, NT=3, column-guarded
        dim3 g((MT_E + 7) / 8, OUTP / 48);
        wmma_gemm<3, false, float, true, true><<<g, blk, 0, stream>>>(
            Tbuf, EDGC, nullptr, nullptr, nullptr, w8h, EDGC, b8, out, OUTC, OUTC, MT_E, EDGC);
    }
}
